// QuantLinear_1726576855767
// MI455X (gfx1250) — matrix operand order repaired and checked by reading, not compiled, not run on hardware
//
#include <hip/hip_runtime.h>

// ---------------------------------------------------------------------------
// GPTQ int4 dequant + GEMM for gfx1250 (MI455X), wave32 + v_wmma_f32_16x16x32_f16
//   out[64,8192] = x[64,8192](fp16) @ dequant(qweight)[8192,8192](fp16) + bias
//
// v4: deeper software pipeline.
//  - qweight words (the HBM stream) are prefetched TWO chunks ahead
//    (w_cur/w_next + issue c+2), so B-dequant never waits on a fresh load.
//  - group constants are derived inside the unrolled body (at chunk c==3,
//    for the NEXT group) and carried in registers across the back-edge, so
//    the group-loop header has no load-wait -> no conservative
//    s_wait_loadcnt 0x0 drain once per group.
//  - A fragments keep 1-chunk lookahead (x is L2-resident, short latency).
// ---------------------------------------------------------------------------

typedef __attribute__((ext_vector_type(16))) _Float16 v16h;
typedef __attribute__((ext_vector_type(8)))  _Float16 v8h;
typedef __attribute__((ext_vector_type(2)))  _Float16 h2;
typedef __attribute__((ext_vector_type(8)))  float    v8f;

constexpr int Mdim  = 64;
constexpr int Kdim  = 8192;
constexpr int Ndim  = 8192;
constexpr int GROUP = 128;          // GPTQ group size
constexpr int MT    = Mdim / 16;    // 4 M-tiles of 16
constexpr int WAVES = 8;            // waves per block (256 threads)

// Dequant one packed half2: t holds two nibbles at bits [3:0] and [7:4].
// 0x6400 | nib is the fp16 encoding of (1024 + nib); the -(1024+z)*s offset
// is pre-folded into zs2, so result = (w - z) * s via one v_pk_fma_f16.
__device__ __forceinline__ h2 dq_pair(unsigned t, h2 s2, h2 zs2) {
    union { unsigned u; h2 h; } c;
    c.u = (t & 0xFu) | ((t & 0xF0u) << 12) | 0x64006400u;
    return c.h * s2 + zs2;   // fp-contract=fast -> v_pk_fma_f16
}

// B fragment: 32x16 fp16, lanes 0-15 = K[0,16), lanes 16-31 = K[16,32).
__device__ __forceinline__ v16h build_B(unsigned w0, unsigned w1, h2 s2, h2 zs2) {
    union { v16h v; h2 p[8]; } B;
#pragma unroll
    for (int j = 0; j < 4; ++j) B.p[j]     = dq_pair(w0 >> (8 * j), s2, zs2);
#pragma unroll
    for (int j = 0; j < 4; ++j) B.p[4 + j] = dq_pair(w1 >> (8 * j), s2, zs2);
    return B.v;
}

// A fragments for one 32-wide K chunk: two b128 loads per M-tile
// (contiguous A layout: lanes 0-15 = K[0,16), lanes 16-31 = K[16,32)).
__device__ __forceinline__ void load_A(const _Float16* __restrict__ base, v16h* A) {
#pragma unroll
    for (int mt = 0; mt < MT; ++mt) {
        union { v16h v; v8h h[2]; } u;
        u.h[0] = *(const v8h*)(base + (size_t)mt * 16 * Kdim);
        u.h[1] = *(const v8h*)(base + (size_t)mt * 16 * Kdim + 8);
        A[mt] = u.v;
    }
}

// Derive the folded per-group constants from raw zero-word + scale.
__device__ __forceinline__ void derive_consts(_Float16 sv, unsigned zraw, int zshift,
                                              h2& s2, h2& zs2) {
    const int      znib = (int)((zraw >> zshift) & 15u);
    const _Float16 zh   = (_Float16)(-(float)sv * (float)(1025 + znib));
    s2.x  = sv; s2.y  = sv;
    zs2.x = zh; zs2.y = zh;
}

template <bool FINAL>
__global__ void __launch_bounds__(WAVES * 32)
gptq_wmma(const _Float16* __restrict__ x,
          const int*      __restrict__ qw,     // [K/8, N] int32
          const int*      __restrict__ qz,     // [G, N/8] int32
          const _Float16* __restrict__ sc,     // [G, N]
          const int*      __restrict__ gidx,   // [K]
          const _Float16* __restrict__ bias,   // [N]
          float*          __restrict__ out,    // FINAL ? [M,N] : [ksplit,M,N]
          int k_per)
{
    const int lane   = threadIdx.x & 31;
    const int wave   = threadIdx.x >> 5;
    const int lane16 = lane & 15;
    const int half_  = lane >> 4;                 // K-half selector
    const int tile_n = blockIdx.x * WAVES + wave;
    const int n      = tile_n * 16 + lane16;      // this lane's output column
    const int kbeg   = blockIdx.y * k_per;
    const int kend   = kbeg + k_per;

    const _Float16* abase  = x + (size_t)lane16 * Kdim + half_ * 16;
    const int       zcol   = n >> 3;
    const int       zshift = (n & 7) * 4;

    v8f zero = {};
    v8f acc[MT];
#pragma unroll
    for (int i = 0; i < MT; ++i) acc[i] = zero;

    // ---- pipeline prologue -------------------------------------------------
    // chunk 0 A fragments, chunk 0 + chunk 1 qweight words in flight
    v16h A[MT];
    load_A(abase + kbeg, A);
    const int rw0 = (kbeg >> 3) + half_ * 2;
    unsigned wc0 = (unsigned)qw[(size_t)rw0 * Ndim + n];          // chunk 0
    unsigned wc1 = (unsigned)qw[(size_t)(rw0 + 1) * Ndim + n];
    unsigned wn0 = (unsigned)qw[(size_t)(rw0 + 4) * Ndim + n];    // chunk 1
    unsigned wn1 = (unsigned)qw[(size_t)(rw0 + 5) * Ndim + n];

    // group 0 constants, derived synchronously once
    h2 s2, zs2;
    {
        const int      g0  = gidx[kbeg];                          // uniform s_load
        const unsigned zr0 = (unsigned)qz[(size_t)g0 * (Ndim / 8) + zcol];
        const _Float16 sv0 = sc[(size_t)g0 * Ndim + n];
        derive_consts(sv0, zr0, zshift, s2, zs2);
    }

    // ---- main loop: 1 group (4 chunks of K=32) per iteration ---------------
    for (int kg = kbeg; kg < kend; kg += GROUP) {
        // Clamped next-group base (shared by the c==3 prefetches).
        const int kgn = (kg + GROUP < kend) ? kg + GROUP : kbeg;

        // Issue next group's raw constant loads now; consumed at c==3.
        const int      gn  = gidx[kgn];
        const unsigned zrn = (unsigned)qz[(size_t)gn * (Ndim / 8) + zcol];
        const _Float16 svn = sc[(size_t)gn * Ndim + n];

#pragma unroll
        for (int c = 0; c < GROUP / 32; ++c) {
            const int k0 = kg + c * 32;
            // Compile-time selects under the unroll -> affine addresses.
            const int kA = (c < 3) ? (k0 + 32) : kgn;                       // c+1
            const int kW = (c < 2) ? (k0 + 64) : ((c == 2) ? kgn : kgn + 32); // c+2

            // ---- issue prefetches FIRST (latency hidden behind WMMAs)
            const int      rn  = (kW >> 3) + half_ * 2;
            const unsigned fw0 = (unsigned)qw[(size_t)rn * Ndim + n];
            const unsigned fw1 = (unsigned)qw[(size_t)(rn + 1) * Ndim + n];
            v16h An[MT];
            load_A(abase + kA, An);

            // ---- dequant B (data loaded 2 chunks ago) + 4 WMMAs
            const v16h B = build_B(wc0, wc1, s2, zs2);
#pragma unroll
            for (int mt = 0; mt < MT; ++mt)
                acc[mt] = __builtin_amdgcn_wmma_f32_16x16x32_f16(
                    false, A[mt], false, B, (short)0, acc[mt], false, false);

            // ---- rotate pipeline registers (SSA-renamed inside the unroll)
            wc0 = wn0; wc1 = wn1;
            wn0 = fw0; wn1 = fw1;
#pragma unroll
            for (int mt = 0; mt < MT; ++mt) A[mt] = An[mt];

            // Derive NEXT group's constants mid-body (precise wait counts,
            // raws have had ~3 chunks in flight). Carried across back-edge.
            if (c == (GROUP / 32) - 1)
                derive_consts(svn, zrn, zshift, s2, zs2);
        }
    }

    // ---- Epilogue: C layout VGPR r -> M = r + 8*half_, lane16 -> N
    float* po = FINAL ? out : (out + (size_t)blockIdx.y * Mdim * Ndim);
    const float bb = FINAL ? (float)bias[n] : 0.0f;
#pragma unroll
    for (int mt = 0; mt < MT; ++mt) {
#pragma unroll
        for (int r = 0; r < 8; ++r) {
            const int m = mt * 16 + half_ * 8 + r;
            float v = acc[mt][r];
            if (FINAL) v = (float)(_Float16)(v + bb);   // round like fp16 output
            po[(size_t)m * Ndim + n] = v;
        }
    }
}

// Sum the 4 K-partials, add bias, round through fp16.
__global__ void __launch_bounds__(256)
gptq_reduce(const float* __restrict__ ws,
            const _Float16* __restrict__ bias,
            float* __restrict__ out)
{
    const int MN = Mdim * Ndim;
    int i = blockIdx.x * blockDim.x + threadIdx.x;
    if (i >= MN) return;
    const int n = i & (Ndim - 1);
    float s = ws[i] + ws[i + MN] + ws[i + 2 * MN] + ws[i + 3 * MN];
    out[i] = (float)(_Float16)(s + (float)bias[n]);
}

extern "C" void kernel_launch(void* const* d_in, const int* in_sizes, int n_in,
                              void* d_out, int out_size, void* d_ws, size_t ws_size,
                              hipStream_t stream)
{
    const _Float16* x    = (const _Float16*)d_in[0];
    const int*      qw   = (const int*)d_in[1];
    const int*      qzr  = (const int*)d_in[2];
    const _Float16* sc   = (const _Float16*)d_in[3];
    const int*      gi   = (const int*)d_in[4];
    const _Float16* bias = (const _Float16*)d_in[5];
    float*          out  = (float*)d_out;

    const size_t MN    = (size_t)Mdim * Ndim;
    const int    tiles = Ndim / 16;               // 512 column tiles
    dim3 block(WAVES * 32);

    if (ws_size >= 4 * MN * sizeof(float)) {
        // K split 4 ways -> 2048 waves, partials in d_ws, small reduce epilogue.
        float* ws = (float*)d_ws;
        dim3 grid(tiles / WAVES, 4);
        gptq_wmma<false><<<grid, block, 0, stream>>>(x, qw, qzr, sc, gi, bias,
                                                     ws, Kdim / 4);
        gptq_reduce<<<(int)((MN + 255) / 256), 256, 0, stream>>>(ws, bias, out);
    } else {
        // Fallback: single split writes final result directly.
        dim3 grid(tiles / WAVES, 1);
        gptq_wmma<true><<<grid, block, 0, stream>>>(x, qw, qzr, sc, gi, bias,
                                                    out, Kdim);
    }
}
